// MultiHeadAttention_42356967473838
// MI455X (gfx1250) — matrix operand order repaired and checked by reading, not compiled, not run on hardware
//
#include <hip/hip_runtime.h>

typedef _Float16 f16;
typedef _Float16 v16h __attribute__((ext_vector_type(16)));
typedef _Float16 v8h  __attribute__((ext_vector_type(8)));
typedef float    v8f  __attribute__((ext_vector_type(8)));

#define SEQ   1024
#define DM    1024
#define NH    16
#define DEPTH 64

// ---- Tensor Data Mover (gfx1250) -------------------------------------------
#if defined(__HIP_DEVICE_COMPILE__) && defined(__gfx1250__) && \
    __has_builtin(__builtin_amdgcn_tensor_load_to_lds) &&      \
    __has_builtin(__builtin_amdgcn_s_wait_tensorcnt)
#define USE_TDM 1
#else
#define USE_TDM 0
#endif

#if USE_TDM
typedef unsigned int v4u __attribute__((ext_vector_type(4)));
typedef int          v4i __attribute__((ext_vector_type(4)));
typedef int          v8i __attribute__((ext_vector_type(8)));

// 2D tile load, 2-byte elements, row-major: LDS gets tile_d1 rows of tile_d0
// elements, rows packed contiguously. Descriptor per CDNA5 ISA D# layout.
// This toolchain exposes the 6-arg builtin: (g0, g1, g2, g3, g4, cpol).
static __device__ __forceinline__ void tdm_load_2d_f16(const void* lds_ptr,
                                                       const void* gptr,
                                                       unsigned tensor_d0,
                                                       unsigned tensor_d1,
                                                       unsigned long long stride0,
                                                       unsigned tile_d0,
                                                       unsigned tile_d1) {
  unsigned long long ga = (unsigned long long)(uintptr_t)gptr;
  unsigned la = (unsigned)(uintptr_t)lds_ptr;  // low 32 bits = LDS byte offset
  v4u g0;
  g0[0] = 1u;                                   // count=1 (valid user D#)
  g0[1] = la;                                   // lds_addr
  g0[2] = (unsigned)(ga & 0xffffffffu);         // global_addr[31:0]
  g0[3] = (unsigned)((ga >> 32) & 0x1ffffffu)   // global_addr[56:32]
          | (2u << 30);                         // type=2 ("image")
  v8i g1;
  g1[0] = (int)(1u << 16);                      // data_size=1 (2 bytes), mask=0
  g1[1] = (int)((tensor_d0 & 0xffffu) << 16);   // tensor_dim0 lo16 @bit48
  g1[2] = (int)((tensor_d0 >> 16) | ((tensor_d1 & 0xffffu) << 16));
  g1[3] = (int)((tensor_d1 >> 16) | (tile_d0 << 16));
  g1[4] = (int)(tile_d1 & 0xffffu);             // tile_dim1; tile_dim2=0
  g1[5] = (int)(unsigned)(stride0 & 0xffffffffu);
  g1[6] = (int)(unsigned)((stride0 >> 32) & 0xffffu);  // stride1=0
  g1[7] = 0;
  v4i z4 = {0, 0, 0, 0};                        // groups 2/3 unused (2D)
  v8i z8 = {0, 0, 0, 0, 0, 0, 0, 0};            // extra group (6-arg form)
  __builtin_amdgcn_tensor_load_to_lds(g0, g1, z4, z4, z8, 0);
}
static __device__ __forceinline__ void tdm_wait() {
  __builtin_amdgcn_s_wait_tensorcnt((short)0);
}
#endif
// ----------------------------------------------------------------------------

static __device__ __forceinline__ v16h pack16(v8h lo, v8h hi) {
  v16h r;
#pragma unroll
  for (int i = 0; i < 8; i++) { r[i] = lo[i]; r[8 + i] = hi[i]; }
  return r;
}

static __device__ __forceinline__ v8f wmma_f16(v16h a, v16h b, v8f c) {
  return __builtin_amdgcn_wmma_f32_16x16x32_f16(false, a, false, b, (short)0, c,
                                                false, false);
}

// ---------------- f32 -> f16 conversion ----------------
__global__ void mha_cvt_f16(const float* __restrict__ src, f16* __restrict__ dst, int n) {
  int i = blockIdx.x * blockDim.x + threadIdx.x;
  if (i < n) dst[i] = (f16)src[i];
}

// ---------------- spatial-bias MLP: relu(relu(m*W1+b1)@W2+b2) ----------------
__global__ void mha_spatial_bias(const float* __restrict__ mdm,
                                 const float* __restrict__ W1, const float* __restrict__ b1,
                                 const float* __restrict__ W2, const float* __restrict__ b2,
                                 float* __restrict__ bias, int n) {
  int i = blockIdx.x * blockDim.x + threadIdx.x;
  if (i >= n) return;
  float m = mdm[i];
  float s = b2[0];
#pragma unroll
  for (int j = 0; j < 16; j++) {
    float h = m * W1[j] + b1[j];
    h = h > 0.f ? h : 0.f;
    s += h * W2[j];
  }
  bias[i] = s > 0.f ? s : 0.f;
}

// ---------------- projection GEMM: C(4096x1024) = A(f16) x W(f16) + bias ----------------
// MODE 0: write f16 head-split (B,H,N,64).  MODE 1: write f32 flat (B,N,1024).
template <int MODE>
__global__ __launch_bounds__(256) void mha_proj_gemm(const f16* __restrict__ A,
                                                     const f16* __restrict__ W,
                                                     const float* __restrict__ bias,
                                                     void* __restrict__ out) {
  __shared__ __align__(16) f16 As[128][32];
  __shared__ __align__(16) f16 Wst[128][32];  // W tile transposed: [n][k]

  const int t = threadIdx.x;
  const int w = t >> 5, lane = t & 31;
  const int wm = w >> 2, wn = w & 3;        // 2x4 wave grid -> 128x128 WG tile
  const int bm = blockIdx.x >> 3, bn = blockIdx.x & 7;
  const int m0 = bm * 128, n0 = bn * 128;
  const int lm = lane & 15, hf = lane >> 4;

  v8f acc[4][2];
#pragma unroll
  for (int a = 0; a < 4; a++)
#pragma unroll
    for (int c = 0; c < 2; c++) acc[a][c] = (v8f){0.f, 0.f, 0.f, 0.f, 0.f, 0.f, 0.f, 0.f};

  for (int kk = 0; kk < DM; kk += 32) {
#if USE_TDM
    // A tile 128x32: one DMA issued by wave 0 (wave-uniform branch; EXEC is
    // ignored by TDM, so other waves must not reach the instruction).
    if (w == 0) {
      tdm_load_2d_f16(&As[0][0], A + (size_t)m0 * DM + kk,
                      /*tensor_d0=*/DM, /*tensor_d1=*/4096, /*stride0=*/DM,
                      /*tile_d0=*/32, /*tile_d1=*/128);
    }
#else
    {  // stage A tile 128x32 (coalesced b128)
      int r = t >> 1, kc = (t & 1) * 16;
      const v8h* gp = (const v8h*)(A + (size_t)(m0 + r) * DM + kk + kc);
      v8h a0 = gp[0], a1 = gp[1];
      *(v8h*)&As[r][kc] = a0;
      *(v8h*)&As[r][kc + 8] = a1;
    }
#endif
    {  // stage W tile 32x128 transposed -> Wst[n][k]
      int k = t >> 3, nc = (t & 7) * 16;
      const f16* gp = W + (size_t)(kk + k) * DM + n0 + nc;
#pragma unroll
      for (int i = 0; i < 16; i++) Wst[nc + i][k] = gp[i];
    }
#if USE_TDM
    if (w == 0) tdm_wait();
#endif
    __syncthreads();

    v16h aA[4], bB[2];
#pragma unroll
    for (int mt = 0; mt < 4; mt++) {
      int m = wm * 64 + mt * 16 + lm;
      aA[mt] = pack16(*(const v8h*)&As[m][hf * 8], *(const v8h*)&As[m][16 + hf * 8]);
    }
#pragma unroll
    for (int nt = 0; nt < 2; nt++) {
      int n = wn * 32 + nt * 16 + lm;
      bB[nt] = pack16(*(const v8h*)&Wst[n][hf * 8], *(const v8h*)&Wst[n][16 + hf * 8]);
    }
#pragma unroll
    for (int mt = 0; mt < 4; mt++)
#pragma unroll
      for (int nt = 0; nt < 2; nt++) acc[mt][nt] = wmma_f16(aA[mt], bB[nt], acc[mt][nt]);
    __syncthreads();
  }

#pragma unroll
  for (int mt = 0; mt < 4; mt++)
#pragma unroll
    for (int nt = 0; nt < 2; nt++)
#pragma unroll
      for (int r = 0; r < 8; r++) {
        int row = m0 + wm * 64 + mt * 16 + r + 8 * hf;
        int col = n0 + wn * 32 + nt * 16 + lm;
        float vv = acc[mt][nt][r] + bias[col];
        if (MODE == 0) {
          int b = row >> 10, nseq = row & 1023, h2 = col >> 6, d = col & 63;
          ((f16*)out)[(((size_t)(b * NH + h2) * SEQ + nseq) << 6) + d] = (f16)vv;
        } else {
          ((float*)out)[(size_t)row * DM + col] = vv;
        }
      }
}

// ---------------- fused logits + bias + softmax + AV ----------------
// block = one (b,h, 16-row q tile); 8 waves partition the 1024 keys (128 each).
__global__ __launch_bounds__(256) void mha_attention(const f16* __restrict__ Qh,
                                                     const f16* __restrict__ Kh,
                                                     const f16* __restrict__ Vh,
                                                     const float* __restrict__ Bias,
                                                     float* __restrict__ attn,
                                                     f16* __restrict__ OhC) {
  __shared__ __align__(16) f16 Qs[16][64];
  __shared__ float red[8][16];
  __shared__ __align__(16) unsigned char UB[32768];  // reused: Ps -> Vs -> Op
  f16(*Ps)[1024] = (f16(*)[1024])UB;                 // 16x1024 f16 = 32KB
  float(*Op)[16][64] = (float(*)[16][64])UB;         // 8x16x64 f32 = 32KB

  const int t = threadIdx.x, w = t >> 5, lane = t & 31;
  const int lm = lane & 15, hf = lane >> 4;
  const int bh = blockIdx.x >> 6, qt = blockIdx.x & 63;
  const int b = bh >> 4, h = bh & 15;
  const int q0 = qt * 16;
  const size_t bhbase = (size_t)bh * SEQ * DEPTH;

  {  // stage 16x64 Q tile
    int row = t >> 4, off = (t & 15) * 4;
    const f16* gp = Qh + bhbase + (size_t)(q0 + row) * DEPTH + off;
    Qs[row][off] = gp[0]; Qs[row][off + 1] = gp[1];
    Qs[row][off + 2] = gp[2]; Qs[row][off + 3] = gp[3];
  }
  __syncthreads();

  v16h aQ[2];
#pragma unroll
  for (int ki = 0; ki < 2; ki++) {
    int k0 = ki * 32;
    aQ[ki] = pack16(*(const v8h*)&Qs[lm][k0 + hf * 8], *(const v8h*)&Qs[lm][k0 + 16 + hf * 8]);
  }

  // ---- logits: 8 column tiles x (depth 64 = 2 WMMA k-steps) ----
  v8f acc[8];
#pragma unroll
  for (int ct = 0; ct < 8; ct++) acc[ct] = (v8f){0.f, 0.f, 0.f, 0.f, 0.f, 0.f, 0.f, 0.f};
#pragma unroll
  for (int ct = 0; ct < 8; ct++) {
    int n = w * 128 + ct * 16 + lm;
    const f16* kp = Kh + bhbase + (size_t)n * DEPTH;
#pragma unroll
    for (int ki = 0; ki < 2; ki++) {
      v16h bK = pack16(*(const v8h*)(kp + ki * 32 + hf * 8),
                       *(const v8h*)(kp + ki * 32 + 16 + hf * 8));
      acc[ct] = wmma_f16(aQ[ki], bK, acc[ct]);
    }
  }

  // ---- scale + spatial bias ----
#pragma unroll
  for (int ct = 0; ct < 8; ct++) {
    int n = w * 128 + ct * 16 + lm;
    const float* bp = Bias + ((size_t)b * SEQ + q0) * SEQ + n;
#pragma unroll
    for (int r = 0; r < 8; r++) acc[ct][r] = acc[ct][r] * 0.125f + bp[(size_t)(r + 8 * hf) * SEQ];
  }

  // ---- softmax: row max ----
  float pm[8];
#pragma unroll
  for (int r = 0; r < 8; r++) {
    float v = acc[0][r];
#pragma unroll
    for (int ct = 1; ct < 8; ct++) v = fmaxf(v, acc[ct][r]);
    pm[r] = v;
  }
#pragma unroll
  for (int off = 1; off < 16; off <<= 1)
#pragma unroll
    for (int r = 0; r < 8; r++) pm[r] = fmaxf(pm[r], __shfl_xor(pm[r], off, 32));
  if (lm == 0)
#pragma unroll
    for (int r = 0; r < 8; r++) red[w][r + 8 * hf] = pm[r];
  __syncthreads();
  float rmax[8];
#pragma unroll
  for (int r = 0; r < 8; r++) {
    float v = -1e30f;
#pragma unroll
    for (int w2 = 0; w2 < 8; w2++) v = fmaxf(v, red[w2][r + 8 * hf]);
    rmax[r] = v;
  }
  __syncthreads();

  // ---- exp + row sum ----
  float ps[8];
#pragma unroll
  for (int r = 0; r < 8; r++) ps[r] = 0.f;
#pragma unroll
  for (int ct = 0; ct < 8; ct++)
#pragma unroll
    for (int r = 0; r < 8; r++) {
      float e = __expf(acc[ct][r] - rmax[r]);
      acc[ct][r] = e;
      ps[r] += e;
    }
#pragma unroll
  for (int off = 1; off < 16; off <<= 1)
#pragma unroll
    for (int r = 0; r < 8; r++) ps[r] += __shfl_xor(ps[r], off, 32);
  if (lm == 0)
#pragma unroll
    for (int r = 0; r < 8; r++) red[w][r + 8 * hf] = ps[r];
  __syncthreads();
  float rinv[8];
#pragma unroll
  for (int r = 0; r < 8; r++) {
    float v = 0.f;
#pragma unroll
    for (int w2 = 0; w2 < 8; w2++) v += red[w2][r + 8 * hf];
    rinv[r] = 1.0f / v;
  }
  __syncthreads();

  // ---- normalize, write attention_weights once, stash P (f16) in LDS ----
#pragma unroll
  for (int ct = 0; ct < 8; ct++) {
    int n = w * 128 + ct * 16 + lm;
    float* ap = attn + ((size_t)bh * SEQ + q0) * SEQ + n;
#pragma unroll
    for (int r = 0; r < 8; r++) {
      int m = r + 8 * hf;
      float p = acc[ct][r] * rinv[r];
      ap[(size_t)m * SEQ] = p;
      Ps[m][n] = (f16)p;
    }
  }
  // A-fragments of P (own wave's 128 columns; same-wave LDS ordering suffices)
  v16h aP[4];
#pragma unroll
  for (int cp = 0; cp < 4; cp++) {
    int base = w * 128 + cp * 32;
    aP[cp] = pack16(*(const v8h*)&Ps[lm][base + hf * 8],
                    *(const v8h*)&Ps[lm][base + 16 + hf * 8]);
  }
  __syncthreads();  // all waves done reading Ps -> UB reusable (fences DScnt)

  // ---- AV: O(16x64) += P(16x128) x V(128x64), V staged through per-wave LDS ----
  f16(*Vs)[64] = (f16(*)[64])(UB + w * 4096);  // 32x64 per-wave chunk
  v8f oacc[4];
#pragma unroll
  for (int dt = 0; dt < 4; dt++) oacc[dt] = (v8f){0.f, 0.f, 0.f, 0.f, 0.f, 0.f, 0.f, 0.f};
  for (int cp = 0; cp < 4; cp++) {
#if USE_TDM
    // per-wave private 32x64 chunk via TDM; rows are stride-64 in memory.
    tdm_load_2d_f16(&Vs[0][0], Vh + bhbase + (size_t)(w * 128 + cp * 32) * DEPTH,
                    /*tensor_d0=*/DEPTH, /*tensor_d1=*/SEQ, /*stride0=*/DEPTH,
                    /*tile_d0=*/DEPTH, /*tile_d1=*/32);
    tdm_wait();
#else
    {  // stage 32 V rows, one row (128B) per lane
      int vr = w * 128 + cp * 32 + lane;
      const v8h* gp = (const v8h*)(Vh + bhbase + (size_t)vr * DEPTH);
      v8h tmp[8];
#pragma unroll
      for (int i = 0; i < 8; i++) tmp[i] = gp[i];
#pragma unroll
      for (int i = 0; i < 8; i++) *(v8h*)&Vs[lane][i * 8] = tmp[i];
    }
#endif
#pragma unroll
    for (int dt = 0; dt < 4; dt++) {
      v16h bV;
#pragma unroll
      for (int j = 0; j < 16; j++) bV[j] = Vs[8 * hf + j + (j >= 8 ? 8 : 0)][dt * 16 + lm];
      oacc[dt] = wmma_f16(aP[cp], bV, oacc[dt]);
    }
  }

  // ---- cross-wave O reduction (Op[w] aliases this wave's own Vs region) ----
#pragma unroll
  for (int dt = 0; dt < 4; dt++)
#pragma unroll
    for (int r = 0; r < 8; r++) Op[w][r + 8 * hf][dt * 16 + lm] = oacc[dt][r];
  __syncthreads();
#pragma unroll
  for (int i = 0; i < 4; i++) {
    int idx = t * 4 + i;
    int m = idx >> 6, d = idx & 63;
    float s = 0.f;
#pragma unroll
    for (int w2 = 0; w2 < 8; w2++) s += Op[w2][m][d];
    OhC[((size_t)b * SEQ + q0 + m) * DM + h * DEPTH + d] = (f16)s;
  }
}

extern "C" void kernel_launch(void* const* d_in, const int* in_sizes, int n_in,
                              void* d_out, int out_size, void* d_ws, size_t ws_size,
                              hipStream_t stream) {
  (void)in_sizes; (void)n_in; (void)out_size; (void)ws_size;
  const float* q = (const float*)d_in[0];
  const float* k = (const float*)d_in[1];
  const float* v = (const float*)d_in[2];
  const float* mdm = (const float*)d_in[3];
  const float* Wq = (const float*)d_in[4];
  const float* bq = (const float*)d_in[5];
  const float* Wk = (const float*)d_in[6];
  const float* bk = (const float*)d_in[7];
  const float* Wv = (const float*)d_in[8];
  const float* bv = (const float*)d_in[9];
  const float* Wo = (const float*)d_in[10];
  const float* bo = (const float*)d_in[11];
  const float* W1 = (const float*)d_in[12];
  const float* b1 = (const float*)d_in[13];
  const float* W2 = (const float*)d_in[14];
  const float* b2 = (const float*)d_in[15];

  char* ws = (char*)d_ws;
  const size_t MB = 1024 * 1024;
  f16* q16 = (f16*)(ws + 0 * MB);
  f16* k16 = (f16*)(ws + 8 * MB);
  f16* v16 = (f16*)(ws + 16 * MB);
  f16* Wq16 = (f16*)(ws + 24 * MB);
  f16* Wk16 = (f16*)(ws + 26 * MB);
  f16* Wv16 = (f16*)(ws + 28 * MB);
  f16* Wo16 = (f16*)(ws + 30 * MB);
  f16* Qh = (f16*)(ws + 32 * MB);
  f16* Kh = (f16*)(ws + 40 * MB);
  f16* Vh = (f16*)(ws + 48 * MB);
  float* Bias = (float*)(ws + 56 * MB);
  f16* OhC = (f16*)(ws + 72 * MB);

  const int NE = 4 * 1024 * 1024;  // q/k/v elements, also B*N*N
  const int NW = 1024 * 1024;      // weight elements
  dim3 blk(256);

  mha_cvt_f16<<<NE / 256, blk, 0, stream>>>(q, q16, NE);
  mha_cvt_f16<<<NE / 256, blk, 0, stream>>>(k, k16, NE);
  mha_cvt_f16<<<NE / 256, blk, 0, stream>>>(v, v16, NE);
  mha_cvt_f16<<<NW / 256, blk, 0, stream>>>(Wq, Wq16, NW);
  mha_cvt_f16<<<NW / 256, blk, 0, stream>>>(Wk, Wk16, NW);
  mha_cvt_f16<<<NW / 256, blk, 0, stream>>>(Wv, Wv16, NW);
  mha_cvt_f16<<<NW / 256, blk, 0, stream>>>(Wo, Wo16, NW);

  mha_spatial_bias<<<NE / 256, blk, 0, stream>>>(mdm, W1, b1, W2, b2, Bias, NE);

  mha_proj_gemm<0><<<256, blk, 0, stream>>>(q16, Wq16, bq, Qh);
  mha_proj_gemm<0><<<256, blk, 0, stream>>>(k16, Wk16, bk, Kh);
  mha_proj_gemm<0><<<256, blk, 0, stream>>>(v16, Wv16, bv, Vh);

  float* attn_out = (float*)d_out + (size_t)4 * 1024 * 1024;
  mha_attention<<<4096, blk, 0, stream>>>(Qh, Kh, Vh, Bias, attn_out, OhC);

  mha_proj_gemm<1><<<256, blk, 0, stream>>>(OhC, Wo16, bo, d_out);
}
